// LNSimpleRNN_29867202576971
// MI455X (gfx1250) — matrix operand order repaired and checked by reading, not compiled, not run on hardware
//
#include <hip/hip_runtime.h>

// ---------------------------------------------------------------------------
// LayerNorm-SimpleRNN, fused persistent kernel for MI455X (gfx1250).
//   B=64, T=512, D=256, H=512.
//   8 persistent workgroups, each owning a 64-wide slice of H.
//   sX = x_t   (64 x 256 bf16, VALU-converted from fp32 each step)
//   sH = h_{t-1} (64 x 512 bf16, staged each step by the Tensor Data Mover)
//   sBt = [Wx;Wh] slice (768 x 64 bf16, K-contiguous, resident all 512 steps)
//   z = [x|h] @ [Wx;Wh] + bias  via v_wmma_f32_16x16x32_bf16 (f32 accumulate)
//   LayerNorm stats exchanged via global scratch + agent-scope grid barrier.
// ---------------------------------------------------------------------------

typedef __attribute__((ext_vector_type(16))) __bf16   v16bf;
typedef __attribute__((ext_vector_type(8)))  float    v8f;
typedef __attribute__((ext_vector_type(4)))  unsigned uv4;
typedef __attribute__((ext_vector_type(8)))  unsigned uv8;

#define BB 64
#define TT 512
#define DD 256
#define HH 512
#define NSLICE 64              // H columns per workgroup
#define NWG (HH / NSLICE)      // 8 workgroups
#define KTOT (DD + HH)         // 768 concatenated K

// LDS strides (elements). All row strides: *2 % 16 == 0 (b128 alignment) and
// (stride*2/4) % 64 == 4 (rows land on distinct banks for the 16-lane frag loads).
#define XSTRIDE 264            // 256 + 8  -> 528 B rows
#define HSTRIDE 520            // 512 + 8  -> 1040 B rows (= TDM pad 16B / 1024B)
#define SBT_STRIDE 776         // 768 + 8  -> 1552 B rows
#define SZ_STRIDE 66

#define SX_BYTES    (BB * XSTRIDE * 2)           // 33792
#define SH_BYTES    (BB * HSTRIDE * 2)           // 66560
#define SBT_BYTES   (NSLICE * SBT_STRIDE * 2)    // 99328
#define SZ_BYTES    (BB * SZ_STRIDE * 4)         // 16896
#define SSTAT_BYTES (BB * 2 * 4)                 // 512

#define SX_OFF   0
#define SH_OFF   (SX_OFF + SX_BYTES)             // 33792
#define SBT_OFF  (SH_OFF + SH_BYTES)             // 100352
#define SZ_OFF   (SBT_OFF + SBT_BYTES)           // 199680
#define SSTAT_OFF (SZ_OFF + SZ_BYTES)            // 216576
#define SMEM_BYTES (SSTAT_OFF + SSTAT_BYTES)     // 217088 < 320KB

// workspace layout (bytes)
#define WS_CTR_OFF   0        // u32 barrier counter (memset to 0 by host)
#define WS_STATS_OFF 1024     // NWG*BB*2 f32 partial LN stats
#define WS_HBUF_OFF  8192     // BB*HH bf16 hidden state

__device__ __forceinline__ void grid_barrier(unsigned* ctr, unsigned target) {
  __syncthreads();
  __builtin_amdgcn_fence(__ATOMIC_RELEASE, "agent");   // publish our stores
  if (threadIdx.x == 0) {
    __hip_atomic_fetch_add(ctr, 1u, __ATOMIC_RELAXED, __HIP_MEMORY_SCOPE_AGENT);
    while (__hip_atomic_load(ctr, __ATOMIC_RELAXED, __HIP_MEMORY_SCOPE_AGENT) < target)
      __builtin_amdgcn_s_sleep(1);
  }
  __syncthreads();
  __builtin_amdgcn_fence(__ATOMIC_ACQUIRE, "agent");   // invalidate stale lines in every wave
}

__device__ __forceinline__ void mma2(const __bf16* aP, const __bf16* b0P, const __bf16* b1P,
                                     v8f& acc0, v8f& acc1) {
  v16bf a;  uint4* ap  = (uint4*)&a;
  ap[0]  = *(const uint4*)(aP);          // A idx 0..7  -> k0..k0+7
  ap[1]  = *(const uint4*)(aP + 8);      // A idx 8..15 -> k0+8..k0+15
  v16bf b0; uint4* bp0 = (uint4*)&b0;
  bp0[0] = *(const uint4*)(b0P);         // B idx 0..15 -> k0..k0+15 (K-contiguous sBt)
  bp0[1] = *(const uint4*)(b0P + 8);
  v16bf b1; uint4* bp1 = (uint4*)&b1;
  bp1[0] = *(const uint4*)(b1P);
  bp1[1] = *(const uint4*)(b1P + 8);
  acc0 = __builtin_amdgcn_wmma_f32_16x16x32_bf16(false, a, false, b0, (short)0, acc0, false, false);
  acc1 = __builtin_amdgcn_wmma_f32_16x16x32_bf16(false, a, false, b1, (short)0, acc1, false, false);
}

__global__ __launch_bounds__(256, 1)
void lnrnn_fused_kernel(const float* __restrict__ inputs,
                        const float* __restrict__ h0,
                        const float* __restrict__ Wx,
                        const float* __restrict__ Wh,
                        const float* __restrict__ bias,
                        const float* __restrict__ gamma,
                        const float* __restrict__ beta,
                        float* __restrict__ out,
                        unsigned* __restrict__ barCtr,
                        float* __restrict__ statsG,
                        __bf16* __restrict__ hbuf) {
  extern __shared__ char smem[];
  __bf16* sX    = (__bf16*)(smem + SX_OFF);
  __bf16* sH    = (__bf16*)(smem + SH_OFF);
  __bf16* sBt   = (__bf16*)(smem + SBT_OFF);    // [n][k], K-contiguous
  float*  sZ    = (float*)(smem + SZ_OFF);
  float*  sStat = (float*)(smem + SSTAT_OFF);

  const int tid   = threadIdx.x;
  const int wg    = blockIdx.x;
  const int nbase = wg * NSLICE;

  // ---- one-time: load weight slice [Wx;Wh](:,nbase:nbase+64) as bf16 -> sBt
  for (int idx = tid; idx < KTOT * NSLICE; idx += 256) {
    const int k = idx >> 6;            // 0..767
    const int n = idx & 63;
    const float w = (k < DD) ? Wx[(size_t)k * HH + nbase + n]
                             : Wh[(size_t)(k - DD) * HH + nbase + n];
    sBt[n * SBT_STRIDE + k] = (__bf16)w;
  }
  // ---- one-time: init hidden-state slice from h0 (bf16)
  for (int idx = tid; idx < BB * NSLICE; idx += 256) {
    const int m = idx >> 6;
    const int n = idx & 63;
    hbuf[m * HH + nbase + n] = (__bf16)h0[m * HH + nbase + n];
  }

  unsigned nbar = 0;
  grid_barrier(barCtr, NWG * ++nbar);

  // ---- Tensor DMA descriptor: hbuf (64 x 1024B, contiguous) -> sH with
  //      16B LDS padding after every 1024B row (HSTRIDE = 520 bf16).
  //      data_size=8B: tensor/tile dim0 = 128 units, dim1 = 64 rows, stride0 = 128.
  const unsigned long long ga = (unsigned long long)hbuf;
  const uv4 tdG0 = { 1u,                                    // count=1, user mode
                     (unsigned)SH_OFF,                      // lds_addr (dyn-LDS base is 0)
                     (unsigned)ga,
                     (unsigned)((ga >> 32) & 0x01FFFFFFu) | 0x80000000u };  // addr[56:32] | type=2
  const uv8 tdG1 = { 0x07D30000u,      // mask=0 | data_size=3(8B) | pad_en | pad_int=7(1024B) | pad_amt=3(16B)
                     128u << 16,       // tensor_dim0 = 128
                     64u << 16,        // tensor_dim1 = 64
                     128u << 16,       // tile_dim0 = 128
                     64u,              // tile_dim1 = 64, tile_dim2 = 0
                     128u,             // tensor_dim0_stride = 128
                     0u, 0u };

  // wave -> output tile mapping: 16 tiles (4 M x 4 N), 8 waves, 2 N-tiles each
  const int lane  = tid & 31;
  const int wave  = tid >> 5;
  const int hs    = lane >> 4;         // half-wave select
  const int l15   = lane & 15;
  const int mtile = wave & 3;
  const int nt0   = (wave >> 2) * 2;

  const __bf16* aX    = sX  + (mtile * 16 + l15) * XSTRIDE;
  const __bf16* aH    = sH  + (mtile * 16 + l15) * HSTRIDE;
  const __bf16* bRow0 = sBt + ((nt0 + 0) * 16 + l15) * SBT_STRIDE;
  const __bf16* bRow1 = sBt + ((nt0 + 1) * 16 + l15) * SBT_STRIDE;
  const float bias0 = bias[nbase + (nt0 + 0) * 16 + l15];
  const float bias1 = bias[nbase + (nt0 + 1) * 16 + l15];

  for (int t = 0; t < TT; ++t) {
    // ---- kick off TDM: h_{t-1} (global bf16) -> sH, async under TENSORcnt
    if (wave == 0) {
      asm volatile("tensor_load_to_lds %0, %1" :: "s"(tdG0), "s"(tdG1) : "memory");
    }
    // ---- overlap: convert x_t fp32 -> bf16 into sX with all 256 threads
    for (int idx = tid; idx < BB * (DD / 4); idx += 256) {
      const int m = idx >> 6;
      const int c = (idx & 63) * 4;
      const float4 v = *(const float4*)(inputs + ((size_t)m * TT + t) * DD + c);
      __bf16* dst = sX + m * XSTRIDE + c;
      dst[0] = (__bf16)v.x; dst[1] = (__bf16)v.y;
      dst[2] = (__bf16)v.z; dst[3] = (__bf16)v.w;
      if (t + 1 < TT)  // warm caches for next timestep (global_prefetch_b8)
        __builtin_prefetch(inputs + ((size_t)m * TT + (t + 1)) * DD + c, 0, 3);
    }
    if (wave == 0) __builtin_amdgcn_s_wait_tensorcnt(0);
    __syncthreads();

    // ---- WMMA: z-tile = [x|h](16x768) @ B(768x16), f32 accumulate
    v8f acc0 = {};
    v8f acc1 = {};
#pragma unroll 4
    for (int kc = 0; kc < DD / 32; ++kc) {              // x part: K = 0..255
      const int ka = kc * 32 + hs * 16;
      const int kb = kc * 32 + hs * 16;
      mma2(aX + ka, bRow0 + kb, bRow1 + kb, acc0, acc1);
    }
#pragma unroll 4
    for (int kc = 0; kc < HH / 32; ++kc) {              // h part: K = 256..767
      const int ka = kc * 32 + hs * 16;
      const int kb = DD + kc * 32 + hs * 16;
      mma2(aH + ka, bRow0 + kb, bRow1 + kb, acc0, acc1);
    }

    // ---- spill z (+bias) to LDS: C layout lane=N, vgpr r=M (+8 upper half-wave)
    const int mrow = mtile * 16 + hs * 8;
#pragma unroll
    for (int r = 0; r < 8; ++r) {
      sZ[(mrow + r) * SZ_STRIDE + (nt0 + 0) * 16 + l15] = acc0[r] + bias0;
      sZ[(mrow + r) * SZ_STRIDE + (nt0 + 1) * 16 + l15] = acc1[r] + bias1;
    }
    __syncthreads();

    // ---- partial LN stats over our 64 columns -> global
    if (tid < BB) {
      float s = 0.f, s2 = 0.f;
      const float* zr = sZ + tid * SZ_STRIDE;
#pragma unroll 8
      for (int n = 0; n < NSLICE; ++n) { const float z = zr[n]; s += z; s2 += z * z; }
      statsG[(wg * BB + tid) * 2 + 0] = s;
      statsG[(wg * BB + tid) * 2 + 1] = s2;
    }
    grid_barrier(barCtr, NWG * ++nbar);

    // ---- reduce all 8 partials -> mu, rstd
    if (tid < BB) {
      float s = 0.f, s2 = 0.f;
#pragma unroll
      for (int w = 0; w < NWG; ++w) {
        s  += statsG[(w * BB + tid) * 2 + 0];
        s2 += statsG[(w * BB + tid) * 2 + 1];
      }
      const float mu = s * (1.0f / HH);
      float var = s2 * (1.0f / HH) - mu * mu;
      var = var < 0.f ? 0.f : var;
      sStat[tid * 2 + 0] = mu;
      sStat[tid * 2 + 1] = rsqrtf(var + 1e-3f);
    }
    __syncthreads();

    // ---- normalize + tanh; write fp32 output and bf16 next-state slice
    for (int idx = tid; idx < BB * NSLICE; idx += 256) {
      const int m  = idx >> 6;
      const int n  = idx & 63;
      const int gn = nbase + n;
      const float z  = sZ[m * SZ_STRIDE + n];
      const float hn = tanhf((z - sStat[m * 2]) * sStat[m * 2 + 1] * gamma[gn] + beta[gn]);
      out[((size_t)m * TT + t) * HH + gn] = hn;
      hbuf[m * HH + gn] = (__bf16)hn;
    }
    grid_barrier(barCtr, NWG * ++nbar);   // h fully published before next step
  }
}

extern "C" void kernel_launch(void* const* d_in, const int* in_sizes, int n_in,
                              void* d_out, int out_size, void* d_ws, size_t ws_size,
                              hipStream_t stream) {
  const float* inputs = (const float*)d_in[0];
  const float* h0     = (const float*)d_in[1];
  const float* Wx     = (const float*)d_in[2];
  const float* Wh     = (const float*)d_in[3];
  const float* bias   = (const float*)d_in[4];
  const float* gamma  = (const float*)d_in[5];
  const float* beta   = (const float*)d_in[6];
  float* out = (float*)d_out;

  unsigned* barCtr = (unsigned*)((char*)d_ws + WS_CTR_OFF);
  float*    statsG = (float*)((char*)d_ws + WS_STATS_OFF);
  __bf16*   hbuf   = (__bf16*)((char*)d_ws + WS_HBUF_OFF);

  // barrier counter must start at 0 every launch (monotonic-target barrier)
  hipMemsetAsync(d_ws, 0, 1024, stream);

  lnrnn_fused_kernel<<<dim3(NWG), dim3(256), SMEM_BYTES, stream>>>(
      inputs, h0, Wx, Wh, bias, gamma, beta, out, barCtr, statsG, hbuf);
}
